// TargetEnhanceModule_59459527246095
// MI455X (gfx1250) — matrix operand order repaired and checked by reading, not compiled, not run on hardware
//
#include <hip/hip_runtime.h>
#include <hip/hip_bf16.h>
#include <math.h>

typedef __bf16 bf16_t;
typedef __attribute__((ext_vector_type(16))) __bf16 v16bf;
typedef __attribute__((ext_vector_type(8)))  __bf16 v8bf;
typedef __attribute__((ext_vector_type(8)))  float  v8f;
typedef __attribute__((ext_vector_type(4)))  float  v4f;

#define BATCH 8
#define CH    64
#define HW    4096
#define BN_EPS 1e-5f

static __device__ __forceinline__ v8f wmma_bf16(v16bf a, v16bf b, v8f c) {
  return __builtin_amdgcn_wmma_f32_16x16x32_bf16(false, a, false, b, (short)0, c,
                                                 false, false);
}

// Build a 16x32 A-operand from two contiguous 8x bf16 chunks.
static __device__ __forceinline__ v16bf load_a16(const bf16_t* p0, const bf16_t* p1) {
  v8bf lo = *(const v8bf*)p0;
  v8bf hi = *(const v8bf*)p1;
  v16bf r;
#pragma unroll
  for (int e = 0; e < 8; ++e) { r[e] = lo[e]; r[8 + e] = hi[e]; }
  return r;
}

// ---------------------------------------------------------------------------
// Kernel 1: fused 1x1 conv + BN for q/k/v.  q gets the saliency mask folded in.
//   Qm[b][i][c], Kb[b][j][c]  (c contiguous, bf16)   Vt[b][c][i] (i contiguous)
// ---------------------------------------------------------------------------
__global__ __launch_bounds__(256) void qkv_kernel(
    const float* __restrict__ x, const float* __restrict__ sal,
    const float* __restrict__ wq, const float* __restrict__ bq,
    const float* __restrict__ gq, const float* __restrict__ beq,
    const float* __restrict__ wk, const float* __restrict__ bk,
    const float* __restrict__ gk, const float* __restrict__ bek,
    const float* __restrict__ wv, const float* __restrict__ bv,
    const float* __restrict__ gv, const float* __restrict__ bev,
    bf16_t* __restrict__ Qm, bf16_t* __restrict__ Kb, bf16_t* __restrict__ Vt)
{
  __shared__ float w[3 * CH * CH];
  __shared__ float pb[3 * CH], ps[3 * CH], pbe[3 * CH];
  const int tid = threadIdx.x;
  for (int k = tid; k < 3 * CH * CH; k += 256) {
    float v = (k < CH * CH) ? wq[k]
            : (k < 2 * CH * CH ? wk[k - CH * CH] : wv[k - 2 * CH * CH]);
    w[k] = v;
  }
  if (tid < CH) {
    const float rs = rsqrtf(1.0f + BN_EPS);
    pb[tid]          = bq[tid]; ps[tid]          = gq[tid] * rs; pbe[tid]          = beq[tid];
    pb[CH + tid]     = bk[tid]; ps[CH + tid]     = gk[tid] * rs; pbe[CH + tid]     = bek[tid];
    pb[2 * CH + tid] = bv[tid]; ps[2 * CH + tid] = gv[tid] * rs; pbe[2 * CH + tid] = bev[tid];
  }
  __syncthreads();

  const int b = blockIdx.x >> 4;
  const int i = ((blockIdx.x & 15) << 8) + tid;

  float xr[CH];
#pragma unroll
  for (int c = 0; c < CH; ++c) xr[c] = x[((size_t)(b * CH + c)) * HW + i];
  const float msk = sal[b * HW + i];

#pragma unroll
  for (int mm = 0; mm < 3; ++mm) {
    const float* wm = &w[mm * CH * CH];
    for (int og = 0; og < CH; og += 8) {
      float acc[8];
#pragma unroll
      for (int oo = 0; oo < 8; ++oo) acc[oo] = pb[mm * CH + og + oo];
      for (int c = 0; c < CH; ++c) {
        const float xv = xr[c];
#pragma unroll
        for (int oo = 0; oo < 8; ++oo)
          acc[oo] = fmaf(wm[(og + oo) * CH + c], xv, acc[oo]);
      }
#pragma unroll
      for (int oo = 0; oo < 8; ++oo)
        acc[oo] = acc[oo] * ps[mm * CH + og + oo] + pbe[mm * CH + og + oo];
      if (mm == 0) {
        v8bf pk;
#pragma unroll
        for (int oo = 0; oo < 8; ++oo) pk[oo] = (bf16_t)(acc[oo] * msk);
        *(v8bf*)&Qm[((size_t)(b * HW + i)) * CH + og] = pk;
      } else if (mm == 1) {
        v8bf pk;
#pragma unroll
        for (int oo = 0; oo < 8; ++oo) pk[oo] = (bf16_t)acc[oo];
        *(v8bf*)&Kb[((size_t)(b * HW + i)) * CH + og] = pk;
      } else {
#pragma unroll
        for (int oo = 0; oo < 8; ++oo)
          Vt[((size_t)(b * CH + og + oo)) * HW + i] = (bf16_t)acc[oo];
      }
    }
  }
}

// ---------------------------------------------------------------------------
// Kernel 2 (pass 1): per-row softmax stats over all j, one wave per 16-row
// i-tile, S = Q^T K via v_wmma_f32_16x16x32_bf16, online (max, sum).
// ---------------------------------------------------------------------------
__global__ __launch_bounds__(256) void stats_kernel(
    const bf16_t* __restrict__ Qm, const bf16_t* __restrict__ Kb,
    float* __restrict__ Mrow, float* __restrict__ Linv)
{
  const int wave = (blockIdx.x << 3) + (threadIdx.x >> 5);
  const int lane = threadIdx.x & 31;
  const int b  = wave >> 8;
  const int it = wave & 255;
  const int i0 = it << 4;
  const int hs  = lane >> 4;
  const int l16 = lane & 15;

  const bf16_t* qrow = Qm + ((size_t)(b * HW) + i0 + l16) * CH;
  const v16bf A0 = load_a16(qrow +      hs * 8, qrow + 16 + hs * 8);
  const v16bf A1 = load_a16(qrow + 32 + hs * 8, qrow + 48 + hs * 8);

  float mrun[8], lrun[8];
#pragma unroll
  for (int r = 0; r < 8; ++r) { mrun[r] = -INFINITY; lrun[r] = 0.0f; }

  const bf16_t* kbase = Kb + (size_t)b * HW * CH;
  for (int jt = 0; jt < 256; ++jt) {
    const bf16_t* krow = kbase + ((size_t)(jt * 16 + l16)) * CH;
    // stream-prefetch 8 tiles ahead (L2 -> near caches); stays inside ws
    __builtin_prefetch(krow + 8 * 16 * CH, 0, 1);
    const v16bf B0 = load_a16(krow +      hs * 8, krow + 16 + hs * 8);
    const v16bf B1 = load_a16(krow + 32 + hs * 8, krow + 48 + hs * 8);
    v8f S = {};
    S = wmma_bf16(A0, B0, S);
    S = wmma_bf16(A1, B1, S);
#pragma unroll
    for (int r = 0; r < 8; ++r) {
      const float s  = S[r];
      const float mn = fmaxf(mrun[r], s);
      lrun[r] = lrun[r] * __expf(mrun[r] - mn) + __expf(s - mn);
      mrun[r] = mn;
    }
  }
  // reduce (m,l) across the 16 lanes that share each row (xor stays in group)
#pragma unroll
  for (int mask = 1; mask <= 8; mask <<= 1) {
#pragma unroll
    for (int r = 0; r < 8; ++r) {
      const float mo = __shfl_xor(mrun[r], mask, 32);
      const float lo = __shfl_xor(lrun[r], mask, 32);
      const float mn = fmaxf(mrun[r], mo);
      lrun[r] = lrun[r] * __expf(mrun[r] - mn) + lo * __expf(mo - mn);
      mrun[r] = mn;
    }
  }
  if (l16 == 0) {
    const int base = b * HW + i0 + hs * 8;
#pragma unroll
    for (int r = 0; r < 8; ++r) {
      Mrow[base + r] = mrun[r];
      Linv[base + r] = 1.0f / lrun[r];
    }
  }
}

// ---------------------------------------------------------------------------
// Kernel 3 (pass 2): one wave per 16-column j-tile.  Recompute S tiles, apply
// softmax, then compute the output TRANSPOSED:  O^T[j, c] = sum_i P^T[j,i] V[i,c]
// The f32 D-layout of the two stacked S tiles IS the A-layout of P^T (16x32),
// so P feeds the next WMMA with only f32->bf16 converts (no lane shuffles).
// V B-operand is two 16B loads from Vt (i0+hs*8 and i0+16+hs*8); epilogue is float4-coalesced.
// ---------------------------------------------------------------------------
__global__ __launch_bounds__(256) void attn_out_kernel(
    const bf16_t* __restrict__ Qm, const bf16_t* __restrict__ Kb,
    const bf16_t* __restrict__ Vt,
    const float* __restrict__ Mrow, const float* __restrict__ Linv,
    const float* __restrict__ x, const float* __restrict__ go,
    const float* __restrict__ beo, float* __restrict__ out)
{
  const int wave = (blockIdx.x << 3) + (threadIdx.x >> 5);
  const int lane = threadIdx.x & 31;
  const int b  = wave >> 8;
  const int jt = wave & 255;
  const int j0 = jt << 4;
  const int hs  = lane >> 4;
  const int l16 = lane & 15;

  const bf16_t* krow = Kb + ((size_t)(b * HW) + j0 + l16) * CH;
  const v16bf BK0 = load_a16(krow +      hs * 8, krow + 16 + hs * 8);
  const v16bf BK1 = load_a16(krow + 32 + hs * 8, krow + 48 + hs * 8);

  v8f accv[4];              // D tiles of O^T: M = j-local, N = c-local (per c-tile)
#pragma unroll
  for (int ct = 0; ct < 4; ++ct) accv[ct] = (v8f){};

  const bf16_t* qbase = Qm + (size_t)b * HW * CH;
  const bf16_t* vbase = Vt + (size_t)b * CH * HW;
  const float*  mb = Mrow + b * HW;
  const float*  lb = Linv + b * HW;

  for (int ic = 0; ic < 128; ++ic) {
    const int i0 = ic * 32;
    const bf16_t* q0 = qbase + ((size_t)(i0 + l16)) * CH;
    const bf16_t* q1 = qbase + ((size_t)(i0 + 16 + l16)) * CH;
    __builtin_prefetch(q0 + 8 * 32 * CH, 0, 1);   // 8 chunks ahead
    const v16bf A00 = load_a16(q0 +      hs * 8, q0 + 16 + hs * 8);
    const v16bf A01 = load_a16(q0 + 32 + hs * 8, q0 + 48 + hs * 8);
    const v16bf A10 = load_a16(q1 +      hs * 8, q1 + 16 + hs * 8);
    const v16bf A11 = load_a16(q1 + 32 + hs * 8, q1 + 48 + hs * 8);
    v8f S0 = {}, S1 = {};
    S0 = wmma_bf16(A00, BK0, S0);
    S0 = wmma_bf16(A01, BK1, S0);
    S1 = wmma_bf16(A10, BK0, S1);
    S1 = wmma_bf16(A11, BK1, S1);

    // row stats for rows (i0 + hs*8 + r) and (i0 + 16 + hs*8 + r)
    const v4f m0a = *(const v4f*)(mb + i0 + hs * 8);
    const v4f m0b = *(const v4f*)(mb + i0 + hs * 8 + 4);
    const v4f l0a = *(const v4f*)(lb + i0 + hs * 8);
    const v4f l0b = *(const v4f*)(lb + i0 + hs * 8 + 4);
    const v4f m1a = *(const v4f*)(mb + i0 + 16 + hs * 8);
    const v4f m1b = *(const v4f*)(mb + i0 + 16 + hs * 8 + 4);
    const v4f l1a = *(const v4f*)(lb + i0 + 16 + hs * 8);
    const v4f l1b = *(const v4f*)(lb + i0 + 16 + hs * 8 + 4);

    // P^T A-operand: element r   <- K = hs*8+r      (tile0 row, own lane)
    //                element 8+r <- K = 16+hs*8+r   (tile1 row, own lane)
    v16bf AP;
#pragma unroll
    for (int r = 0; r < 4; ++r) {
      AP[r]      = (bf16_t)(__expf(S0[r]     - m0a[r]) * l0a[r]);
      AP[r + 4]  = (bf16_t)(__expf(S0[r + 4] - m0b[r]) * l0b[r]);
      AP[r + 8]  = (bf16_t)(__expf(S1[r]     - m1a[r]) * l1a[r]);
      AP[r + 12] = (bf16_t)(__expf(S1[r + 4] - m1b[r]) * l1b[r]);
    }

    // B operand per c-tile: e<8 <- V[i0 + hs*8 + e], e>=8 <- V[i0 + 16 + hs*8 + e-8] (c = ct*16 + l16)
#pragma unroll
    for (int ct = 0; ct < 4; ++ct) {
      const bf16_t* vp = vbase + ((size_t)(ct * 16 + l16)) * HW + i0;
      const v16bf BV = load_a16(vp + hs * 8, vp + 16 + hs * 8);
      accv[ct] = wmma_bf16(AP, BV, accv[ct]);
    }
  }

  // epilogue: O^T D tile -> each lane owns channel (ct*16+l16), 8 contiguous j.
  const float rs = rsqrtf(1.0f + BN_EPS);
#pragma unroll
  for (int ct = 0; ct < 4; ++ct) {
    const int ch = ct * 16 + l16;
    const float sc = go[ch] * rs;
    const float bb = beo[ch];
    const size_t base = ((size_t)(b * CH + ch)) * HW + j0 + hs * 8;
    const v4f xa = *(const v4f*)(x + base);
    const v4f xb = *(const v4f*)(x + base + 4);
    v4f oa, ob;
#pragma unroll
    for (int r = 0; r < 4; ++r) {
      oa[r] = (accv[ct][r]     + xa[r]) * sc + bb;
      ob[r] = (accv[ct][r + 4] + xb[r]) * sc + bb;
    }
    *(v4f*)(out + base)     = oa;
    *(v4f*)(out + base + 4) = ob;
  }
}

extern "C" void kernel_launch(void* const* d_in, const int* in_sizes, int n_in,
                              void* d_out, int out_size, void* d_ws, size_t ws_size,
                              hipStream_t stream) {
  const float* x   = (const float*)d_in[0];
  const float* sal = (const float*)d_in[1];
  const float* wq  = (const float*)d_in[2];
  const float* bq  = (const float*)d_in[3];
  const float* gq  = (const float*)d_in[4];
  const float* beq = (const float*)d_in[5];
  const float* wk  = (const float*)d_in[6];
  const float* bk  = (const float*)d_in[7];
  const float* gk  = (const float*)d_in[8];
  const float* bek = (const float*)d_in[9];
  const float* wv  = (const float*)d_in[10];
  const float* bv  = (const float*)d_in[11];
  const float* gv  = (const float*)d_in[12];
  const float* bev = (const float*)d_in[13];
  const float* go  = (const float*)d_in[14];
  const float* beo = (const float*)d_in[15];

  // workspace: Qm/Kb/Vt bf16 (3 * 4 MiB) + Mrow/Linv f32 (2 * 128 KiB)
  char* ws = (char*)d_ws;
  bf16_t* Qm = (bf16_t*)ws;
  bf16_t* Kb = Qm + (size_t)BATCH * HW * CH;
  bf16_t* Vt = Kb + (size_t)BATCH * HW * CH;
  float* Mrow = (float*)(Vt + (size_t)BATCH * HW * CH);
  float* Linv = Mrow + (size_t)BATCH * HW;

  qkv_kernel<<<dim3(BATCH * (HW / 256)), dim3(256), 0, stream>>>(
      x, sal, wq, bq, gq, beq, wk, bk, gk, bek, wv, bv, gv, bev, Qm, Kb, Vt);
  stats_kernel<<<dim3((BATCH * (HW / 16)) / 8), dim3(256), 0, stream>>>(
      Qm, Kb, Mrow, Linv);
  attn_out_kernel<<<dim3((BATCH * (HW / 16)) / 8), dim3(256), 0, stream>>>(
      Qm, Kb, Vt, Mrow, Linv, x, go, beo, (float*)d_out);
}
